// MPNNLayer_7370163880383
// MI455X (gfx1250) — matrix operand order repaired and checked by reading, not compiled, not run on hardware
//
#include <hip/hip_runtime.h>

// ---------------------------------------------------------------------------
// MPNN layer for MI455X (gfx1250): bf16 WMMA pipeline.
//   H=256, NUM_IN=512, K=30, d_in=768.  M rows = B*N*K.
// ---------------------------------------------------------------------------

typedef __bf16 v16bf __attribute__((ext_vector_type(16)));
typedef __bf16 v8bf  __attribute__((ext_vector_type(8)));
typedef __bf16 v4bf  __attribute__((ext_vector_type(4)));
typedef float  v8f   __attribute__((ext_vector_type(8)));

#define NB     4          // (b,n) nodes per workgroup
#define KNBR   30         // neighbors per node
#define ROWS   (NB * 32)  // 128 M-rows (30 real + 2 pad per node)
#define DIN    768
#define HDIM   256
#define LDE    776        // DIN + 8 bf16 pad -> breaks 64-bank alias
#define LDH    264        // HDIM + 8 bf16 pad

#define EIN_B  (ROWS * LDE * 2)   // 198656
#define H1_B   (ROWS * LDH * 2)   //  67584
#define HV_B   (NB * HDIM * 2)    //   2048
#define MSK_B  (ROWS * 4)         //    512
#define SMEM_B (EIN_B + H1_B + HV_B + MSK_B)  // 268800 <= 320KB

// ---- weight transpose + f32->bf16 conversion: dst[h*R + d] = src[d*C + h]
__global__ void wcvt_kernel(const float* __restrict__ src,
                            __bf16* __restrict__ dst, int R, int C) {
  int i = blockIdx.x * blockDim.x + threadIdx.x;
  if (i >= R * C) return;
  int d = i / C;
  int h = i - d * C;
  dst[h * R + d] = (__bf16)src[i];
}

// ---- A fragment: 16-bit 16x32, per ISA: lane<16 holds K {0..7,16..23},
//      lane>=16 holds K {8..15,24..31}, row M = lane&15. Two b128 LDS loads.
__device__ __forceinline__ v16bf load_a(const __bf16* A, int lda,
                                        int mrow0, int k0, int lane) {
  const int half = lane >> 4, l15 = lane & 15;
  const __bf16* p = A + (mrow0 + l15) * lda + k0 + 8 * half;
  v8bf a0 = *(const v8bf*)p;
  v8bf a1 = *(const v8bf*)(p + 16);
  return __builtin_shufflevector(a0, a1, 0,1,2,3,4,5,6,7,8,9,10,11,12,13,14,15);
}

// ---- B fragment: 16-bit 32x16, lane&15 = column N, lane<16 holds K {0..7,16..23},
//      lane>=16 holds K {8..15,24..31} (matches A); weights stored transposed bf16 [h][d].
__device__ __forceinline__ v16bf load_b(const __bf16* Wt, int ldb,
                                        int h0, int k0, int lane) {
  const int half = lane >> 4, l15 = lane & 15;
  const __bf16* p = Wt + (size_t)(h0 + l15) * ldb + k0 + 8 * half;
  v8bf b0 = *(const v8bf*)p;
  v8bf b1 = *(const v8bf*)(p + 16);
  return __builtin_shufflevector(b0, b1, 0,1,2,3,4,5,6,7,8,9,10,11,12,13,14,15);
}

// ---- 128 x 32 output strip through one layer: acc[mt][nt] 16x16 f32 tiles.
__device__ __forceinline__ void gemm_strip(const __bf16* __restrict__ A, int lda,
                                           const __bf16* __restrict__ Wt, int ldb,
                                           int Kdim, int n0, int lane,
                                           v8f (&acc)[8][2]) {
  for (int k0 = 0; k0 < Kdim; k0 += 32) {
    v16bf bf0 = load_b(Wt, ldb, n0,      k0, lane);
    v16bf bf1 = load_b(Wt, ldb, n0 + 16, k0, lane);
#pragma unroll
    for (int mt = 0; mt < 8; ++mt) {
      v16bf af = load_a(A, lda, mt * 16, k0, lane);
      acc[mt][0] = __builtin_amdgcn_wmma_f32_16x16x32_bf16(
          false, af, false, bf0, (short)0, acc[mt][0], false, false);
      acc[mt][1] = __builtin_amdgcn_wmma_f32_16x16x32_bf16(
          false, af, false, bf1, (short)0, acc[mt][1], false, false);
    }
  }
}

__device__ __forceinline__ void zero_acc(v8f (&acc)[8][2]) {
  v8f z = {0.f, 0.f, 0.f, 0.f, 0.f, 0.f, 0.f, 0.f};
#pragma unroll
  for (int mt = 0; mt < 8; ++mt) { acc[mt][0] = z; acc[mt][1] = z; }
}

// ---- bias + relu + bf16 store to LDS (C/D layout: vgpr r -> M = 16mt+r+8*half,
//      N = lane&15)
__device__ __forceinline__ void epi_relu(const v8f (&acc)[8][2],
                                         const float* __restrict__ bias,
                                         __bf16* __restrict__ H, int ldh,
                                         int n0, int lane) {
  const int half = lane >> 4, l15 = lane & 15;
#pragma unroll
  for (int nt = 0; nt < 2; ++nt) {
    const int col = n0 + nt * 16 + l15;
    const float bv = bias[col];
#pragma unroll
    for (int mt = 0; mt < 8; ++mt) {
#pragma unroll
      for (int r = 0; r < 8; ++r) {
        float v = acc[mt][nt][r] + bv;
        v = v > 0.f ? v : 0.f;
        H[(mt * 16 + r + 8 * half) * ldh + col] = (__bf16)v;
      }
    }
  }
}

__global__ __launch_bounds__(256) void mpnn_kernel(
    const float* __restrict__ h_V, const float* __restrict__ h_E,
    const float* __restrict__ mask,
    const float* __restrict__ b1, const float* __restrict__ b2,
    const float* __restrict__ b3,
    const __bf16* __restrict__ w1t, const __bf16* __restrict__ w2t,
    const __bf16* __restrict__ w3t,
    float* __restrict__ out) {
  __shared__ __align__(16) unsigned char smem[SMEM_B];
  __bf16* Ein = (__bf16*)smem;                    // [128][776] bf16 input
  __bf16* H1  = (__bf16*)(smem + EIN_B);          // [128][264] bf16
  __bf16* H2  = Ein;                              // aliases Ein (dead after L1)
  __bf16* hv  = (__bf16*)(smem + EIN_B + H1_B);   // [4][256] bf16 h_V
  float*  msk = (float*)(smem + EIN_B + H1_B + HV_B);  // [128] mask/30

  const int tid  = threadIdx.x;
  const int lane = tid & 31;
  const int wave = tid >> 5;
  const int p0   = blockIdx.x * NB;   // first flattened (b,n) node

  // ---------------- stage 0: stage inputs into LDS as bf16 ----------------
  if (tid < ROWS) {                   // mask / 30, zero for pad rows
    int g = tid >> 5, k = tid & 31;
    float m = 0.f;
    if (k < KNBR) m = mask[(size_t)(p0 + g) * KNBR + k] * (1.0f / 30.0f);
    msk[tid] = m;
  }
  {                                   // h_V -> bf16 scratch (4 elems/thread)
    int g = tid >> 6;
    int c = (tid & 63) * 4;
    float4 v = *(const float4*)(h_V + (size_t)(p0 + g) * HDIM + c);
    __bf16* d = hv + g * HDIM + c;
    d[0] = (__bf16)v.x; d[1] = (__bf16)v.y; d[2] = (__bf16)v.z; d[3] = (__bf16)v.w;
  }
#pragma unroll 4
  for (int it = 0; it < 60; ++it) {   // h_E -> Ein cols 256..767
    int q   = it * 256 + tid;         // 0..15359 = 4 nodes * 30 k * 128 quads
    int g   = q / 3840;
    int rem = q - g * 3840;
    int k   = rem >> 7;
    int cq  = rem & 127;
    float4 v = *(const float4*)(h_E + (((size_t)(p0 + g) * KNBR + k) << 9) + cq * 4);
    v4bf bv = {(__bf16)v.x, (__bf16)v.y, (__bf16)v.z, (__bf16)v.w};
    *(v4bf*)(Ein + (g * 32 + k) * LDE + HDIM + cq * 4) = bv;
  }
  for (int idx = tid; idx < 8 * 130; idx += 256) {  // zero pad rows (k=30,31)
    int rowi = idx / 130;
    int c4   = HDIM + (idx - rowi * 130) * 4;
    int row  = (rowi >> 1) * 32 + 30 + (rowi & 1);
    v4bf z = {};
    *(v4bf*)(Ein + row * LDE + c4) = z;
  }
  __syncthreads();
#pragma unroll 4
  for (int it = 0; it < 32; ++it) {   // broadcast h_V into Ein cols 0..255
    int idx = it * 256 + tid;         // 128 rows * 64 quads
    int row = idx >> 6;
    int cq  = (idx & 63) * 4;
    v4bf v = *(const v4bf*)(hv + (row >> 5) * HDIM + cq);
    *(v4bf*)(Ein + row * LDE + cq) = v;
  }
  __syncthreads();

  // ---------------- three bf16-WMMA layers ----------------
  const int n0 = wave * 32;
  v8f acc[8][2];

  zero_acc(acc);
  gemm_strip(Ein, LDE, w1t, DIN, DIN, n0, lane, acc);   // layer 1, K=768
  epi_relu(acc, b1, H1, LDH, n0, lane);
  __syncthreads();

  zero_acc(acc);
  gemm_strip(H1, LDH, w2t, HDIM, HDIM, n0, lane, acc);  // layer 2, K=256
  epi_relu(acc, b2, H2, LDH, n0, lane);
  __syncthreads();

  zero_acc(acc);
  gemm_strip(H2, LDH, w3t, HDIM, HDIM, n0, lane, acc);  // layer 3, K=256

  // ---------------- masked mean over neighbors + residual ----------------
  {
    const int half = lane >> 4, l15 = lane & 15;
#pragma unroll
    for (int nt = 0; nt < 2; ++nt) {
      const float bv = b3[n0 + nt * 16 + l15];
#pragma unroll
      for (int g = 0; g < NB; ++g) {
        float s = 0.f;
#pragma unroll
        for (int t = 0; t < 2; ++t) {
          const int mt = 2 * g + t;
#pragma unroll
          for (int r = 0; r < 8; ++r) {
            const int row = mt * 16 + r + 8 * half;
            s += msk[row] * (acc[mt][nt][r] + bv);
          }
        }
        s += __shfl_xor(s, 16, 32);   // join the two lane halves (wave32)
        if (lane < 16) {
          size_t idx = (size_t)(p0 + g) * HDIM + n0 + nt * 16 + lane;
          out[idx] = h_V[idx] + s;
        }
      }
    }
  }
}

// ---------------------------------------------------------------------------
extern "C" void kernel_launch(void* const* d_in, const int* in_sizes, int n_in,
                              void* d_out, int out_size, void* d_ws, size_t ws_size,
                              hipStream_t stream) {
  const float* h_V  = (const float*)d_in[0];
  const float* h_E  = (const float*)d_in[1];
  const float* mask = (const float*)d_in[2];
  const float* W1   = (const float*)d_in[3];
  const float* b1   = (const float*)d_in[4];
  const float* W2   = (const float*)d_in[5];
  const float* b2   = (const float*)d_in[6];
  const float* W3   = (const float*)d_in[7];
  const float* b3   = (const float*)d_in[8];
  float* out = (float*)d_out;

  // workspace: transposed bf16 weights (w1t [256][768], w2t/w3t [256][256])
  __bf16* w1t = (__bf16*)d_ws;
  __bf16* w2t = (__bf16*)((char*)d_ws + (size_t)DIN * HDIM * 2);
  __bf16* w3t = (__bf16*)((char*)d_ws + (size_t)DIN * HDIM * 2 + (size_t)HDIM * HDIM * 2);

  wcvt_kernel<<<(DIN * HDIM + 255) / 256, 256, 0, stream>>>(W1, w1t, DIN, HDIM);
  wcvt_kernel<<<(HDIM * HDIM + 255) / 256, 256, 0, stream>>>(W2, w2t, HDIM, HDIM);
  wcvt_kernel<<<(HDIM * HDIM + 255) / 256, 256, 0, stream>>>(W3, w3t, HDIM, HDIM);

  const int pairs = in_sizes[0] / HDIM;   // B*N flattened nodes (= 8000)
  mpnn_kernel<<<pairs / NB, 256, 0, stream>>>(h_V, h_E, mask, b1, b2, b3,
                                              w1t, w2t, w3t, out);
}
